// PointNetPPClassification_9947144257776
// MI455X (gfx1250) — matrix operand order repaired and checked by reading, not compiled, not run on hardware
//
#include <hip/hip_runtime.h>
#include <hip/hip_bf16.h>
#include <math.h>

// ---------------------------------------------------------------------------
// PointNet++ (SSG) classification forward for MI455X / gfx1250.
// Channel-mixing layers run as wave32 WMMA GEMMs (v_wmma_f32_16x16x32_f16),
// f32 accumulate, bias+BN+ReLU fused in the epilogue.
//  - Activations are stored with K padded to multiples of 32 (zero-filled) so
//    the GEMM inner loop does unconditional float4 (b128) loads: no branches.
//  - Weights are pre-packed per launch into the WMMA B-fragment layout in f16
//    ([ntile][kchunk][lane][16 halves]) so the B fragment is one contiguous
//    32-byte load per lane.
// Geometry kernels (FPS / ball-query / grouping / max-pool / softmax) are
// plain fp32 VALU - they are bandwidth-trivial.
// ---------------------------------------------------------------------------

typedef __attribute__((ext_vector_type(16))) _Float16 v16h;
typedef __attribute__((ext_vector_type(8)))  float    v8f;

#define BN_RSQRT 0.99999500003749971875f  // 1/sqrt(1 + 1e-5)

static __host__ __device__ inline int align32i(int v) { return (v + 31) & ~31; }

// ---------------------------------------------------------------------------
// Farthest point sampling: one block per batch, serial over npoint steps,
// LDS tree argmax (first-occurrence tie-break => smallest index).
// ---------------------------------------------------------------------------
__global__ void fps_kernel(const float* __restrict__ xyz, int* __restrict__ idx,
                           int N, int S) {
  __shared__ float dist[4096];
  __shared__ float rv[1024];
  __shared__ int   ri[1024];
  __shared__ int   cur;

  const int b  = blockIdx.x;
  const int t  = threadIdx.x;
  const int nt = blockDim.x;
  const float* p = xyz + (size_t)b * N * 3;

  for (int i = t; i < N; i += nt) dist[i] = 1e10f;
  if (t == 0) { cur = 0; idx[(size_t)b * S] = 0; }
  __syncthreads();

  for (int it = 1; it < S; ++it) {
    const int c  = cur;
    const float cx = p[c * 3 + 0], cy = p[c * 3 + 1], cz = p[c * 3 + 2];
    float best = -1.0f; int bi = 0;
    for (int i = t; i < N; i += nt) {
      const float dx = p[i * 3 + 0] - cx;
      const float dy = p[i * 3 + 1] - cy;
      const float dz = p[i * 3 + 2] - cz;
      const float d  = dx * dx + dy * dy + dz * dz;
      const float dd = fminf(dist[i], d);
      dist[i] = dd;
      if (dd > best) { best = dd; bi = i; }
    }
    rv[t] = best; ri[t] = bi;
    __syncthreads();
    for (int s = nt >> 1; s > 0; s >>= 1) {
      if (t < s) {
        if (rv[t + s] > rv[t] || (rv[t + s] == rv[t] && ri[t + s] < ri[t])) {
          rv[t] = rv[t + s]; ri[t] = ri[t + s];
        }
      }
      __syncthreads();
    }
    if (t == 0) { cur = ri[0]; idx[(size_t)b * S + it] = cur; }
    __syncthreads();
  }
}

// out[g] = xyz[b, idx[g], :]
__global__ void gather_xyz_kernel(const float* __restrict__ xyz,
                                  const int* __restrict__ idx,
                                  float* __restrict__ out,
                                  int N, int S, int total) {
  const int g = blockIdx.x * blockDim.x + threadIdx.x;
  if (g >= total) return;
  const int b  = g / S;
  const int pi = idx[g];
  const float* p = xyz + ((size_t)b * N + pi) * 3;
  out[(size_t)g * 3 + 0] = p[0];
  out[(size_t)g * 3 + 1] = p[1];
  out[(size_t)g * 3 + 2] = p[2];
}

// Ball query: first nsample indices (ascending) with sqdist <= r2, padded
// with the first hit (matches reference sort + pad-with-first).
__global__ void query_ball_kernel(const float* __restrict__ xyz,
                                  const float* __restrict__ new_xyz,
                                  int* __restrict__ gidx,
                                  int N, int S, int nsample, float r2, int total) {
  const int g = blockIdx.x * blockDim.x + threadIdx.x;
  if (g >= total) return;
  const int b = g / S;
  const float* p = xyz + (size_t)b * N * 3;
  const float cx = new_xyz[(size_t)g * 3 + 0];
  const float cy = new_xyz[(size_t)g * 3 + 1];
  const float cz = new_xyz[(size_t)g * 3 + 2];
  int* out = gidx + (size_t)g * nsample;
  int cnt = 0;
  for (int n = 0; n < N; ++n) {
    const float dx = p[n * 3 + 0] - cx;
    const float dy = p[n * 3 + 1] - cy;
    const float dz = p[n * 3 + 2] - cz;
    if (dx * dx + dy * dy + dz * dz <= r2) {
      out[cnt++] = n;
      if (cnt == nsample) break;
    }
  }
  const int f = (cnt > 0) ? out[0] : 0;
  for (int k = cnt; k < nsample; ++k) out[k] = f;
}

// feats row (stride Cpad, zero tail): [xyz - centroid | gathered points | 0s]
__global__ void group_feats_kernel(const float* __restrict__ xyz,
                                   const float* __restrict__ new_xyz,
                                   const int* __restrict__ gidx,
                                   const float* __restrict__ points,
                                   float* __restrict__ feats,
                                   int N, int S, int ns, int Cf, int Cpad,
                                   int total) {
  const int g = blockIdx.x * blockDim.x + threadIdx.x;
  if (g >= total) return;
  const int bs = g / ns;
  const int b  = bs / S;
  const int pi = gidx[g];
  const float* p = xyz + ((size_t)b * N + pi) * 3;
  const float* c = new_xyz + (size_t)bs * 3;
  float* o = feats + (size_t)g * Cpad;
  o[0] = p[0] - c[0];
  o[1] = p[1] - c[1];
  o[2] = p[2] - c[2];
  if (points) {
    const float* q = points + ((size_t)b * N + pi) * Cf;
    for (int cc = 0; cc < Cf; ++cc) o[3 + cc] = q[cc];
  }
  for (int cc = 3 + Cf; cc < Cpad; ++cc) o[cc] = 0.0f;
}

// out row (stride Cpad): [xyz | pts | 0s]   (SA3 global concat)
__global__ void concat3_kernel(const float* __restrict__ xyz,
                               const float* __restrict__ pts,
                               float* __restrict__ out, int total, int C, int Cpad) {
  const int g = blockIdx.x * blockDim.x + threadIdx.x;
  if (g >= total) return;
  float* o = out + (size_t)g * Cpad;
  o[0] = xyz[(size_t)g * 3 + 0];
  o[1] = xyz[(size_t)g * 3 + 1];
  o[2] = xyz[(size_t)g * 3 + 2];
  const float* q = pts + (size_t)g * C;
  for (int c = 0; c < C; ++c) o[3 + c] = q[c];
  for (int c = 3 + C; c < Cpad; ++c) o[c] = 0.0f;
}

// ---------------------------------------------------------------------------
// Weight packing: W (N x K, f32 row-major) -> f16 WMMA B-fragments, padded to
// N16 = ceil(N/16) tiles and KC = ceil(K/32) chunks (zero fill).
// Element t = ((ntile*KC + kc)*32 + lane)*16 + j holds W[n][k] with
//   n = ntile*16 + (lane & 15),  k = kc*32 + (lane>=16 ? 8 : 0) + (j < 8 ? j : j + 8).
// ---------------------------------------------------------------------------
__global__ void pack_weights_kernel(const float* __restrict__ W,
                                    _Float16* __restrict__ Bp,
                                    int N, int K, int N16, int KC) {
  const long long total = (long long)N16 * KC * 32 * 16;
  const long long t = (long long)blockIdx.x * blockDim.x + threadIdx.x;
  if (t >= total) return;
  const int j     = (int)(t & 15);
  const int lane  = (int)((t >> 4) & 31);
  const long long rem = t >> 9;
  const int kc    = (int)(rem % KC);
  const int nt    = (int)(rem / KC);
  const int n = nt * 16 + (lane & 15);
  const int k = kc * 32 + ((lane >= 16) ? 8 : 0) + (j < 8 ? j : j + 8);
  _Float16 v = (_Float16)0.0f;
  if (n < N && k < K) v = (_Float16)W[(size_t)n * K + k];
  Bp[t] = v;
}

// ---------------------------------------------------------------------------
// Fused WMMA GEMM: out[M,Nout] = act( bn( A[M,Kpad] * Wpacked^T + bias ) )
// A: f32, row stride Kpad (Kpad % 32 == 0, zero padded). One wave per 16x16
// output tile. Inner loop: 4x float4 loads (A) + one 32B load (B fragment),
// all unconditional, then one v_wmma_f32_16x16x32_f16.
// ---------------------------------------------------------------------------
__global__ void gemm_wmma(const float* __restrict__ A,
                          const _Float16* __restrict__ Bp,
                          const float* __restrict__ bias,
                          const float* __restrict__ gamma,
                          const float* __restrict__ beta,
                          float* __restrict__ out,
                          int M, int Nout, int Kpad, int fuse) {
  const int wave  = threadIdx.x >> 5;
  const int lane  = threadIdx.x & 31;
  const int tileM = (blockIdx.y * 4 + wave) * 16;
  const int ntile = blockIdx.x;
  if (tileM >= M) return;  // wave-uniform: EXEC all-ones around WMMA

  const int  ml = lane & 15;
  const bool hi = lane >= 16;
  const int  KC = Kpad >> 5;
  const int  h4 = hi ? 2 : 0;  // float4 offset for this lane-half

  const float4* Ar = (const float4*)(A + (size_t)(tileM + ml) * Kpad);
  const v16h*   Bt = (const v16h*)Bp + (size_t)ntile * KC * 32 + lane;

  v8f acc = {};
  for (int kc = 0; kc < KC; ++kc) {
    const float4 f0 = Ar[kc * 8 + h4 + 0];  // K = kc*32 + h8 + 0..3
    const float4 f1 = Ar[kc * 8 + h4 + 1];  // K = kc*32 + h8 + 4..7
    const float4 f2 = Ar[kc * 8 + h4 + 4];  // K = kc*32 + 16 + h8 + 0..3
    const float4 f3 = Ar[kc * 8 + h4 + 5];  // K = kc*32 + 16 + h8 + 4..7
    const v16h   bf = Bt[(size_t)kc * 32];

    v16h af;
    af[0]  = (_Float16)f0.x; af[1]  = (_Float16)f0.y;
    af[2]  = (_Float16)f0.z; af[3]  = (_Float16)f0.w;
    af[4]  = (_Float16)f1.x; af[5]  = (_Float16)f1.y;
    af[6]  = (_Float16)f1.z; af[7]  = (_Float16)f1.w;
    af[8]  = (_Float16)f2.x; af[9]  = (_Float16)f2.y;
    af[10] = (_Float16)f2.z; af[11] = (_Float16)f2.w;
    af[12] = (_Float16)f3.x; af[13] = (_Float16)f3.y;
    af[14] = (_Float16)f3.z; af[15] = (_Float16)f3.w;

    acc = __builtin_amdgcn_wmma_f32_16x16x32_f16(
        /*neg_a=*/false, af, /*neg_b=*/false, bf,
        /*c_mod=*/(short)0, acc, /*reuse_a=*/false, /*reuse_b=*/false);
  }

  const int ncol = ntile * 16 + ml;
  if (ncol < Nout) {
    const float bb = bias[ncol];
    float g = 1.0f, be = 0.0f;
    if (fuse) { g = gamma[ncol] * BN_RSQRT; be = beta[ncol]; }
    const int mbase = tileM + (hi ? 8 : 0);
#pragma unroll
    for (int r = 0; r < 8; ++r) {
      float v = acc[r] + bb;
      if (fuse) { v = fmaxf(v * g + be, 0.0f); }
      out[(size_t)(mbase + r) * Nout + ncol] = v;
    }
  }
}

// out[g, c] = max_k in[(g*ns + k), c]
__global__ void rowmax_kernel(const float* __restrict__ in, float* __restrict__ out,
                              int G, int ns, int C) {
  const long long gid = (long long)blockIdx.x * blockDim.x + threadIdx.x;
  if (gid >= (long long)G * C) return;
  const int g = (int)(gid / C);
  const int c = (int)(gid % C);
  const float* p = in + ((size_t)g * ns) * C + c;
  float m = p[0];
  for (int k = 1; k < ns; ++k) m = fmaxf(m, p[(size_t)k * C]);
  out[gid] = m;
}

__global__ void logsoftmax_kernel(const float* __restrict__ in, float* __restrict__ out,
                                  int Bn, int C) {
  const int b = blockIdx.x * blockDim.x + threadIdx.x;
  if (b >= Bn) return;
  const float* l = in + (size_t)b * C;
  float m = l[0];
  for (int c = 1; c < C; ++c) m = fmaxf(m, l[c]);
  float s = 0.0f;
  for (int c = 0; c < C; ++c) s += expf(l[c] - m);
  const float ls = logf(s);
  for (int c = 0; c < C; ++c) out[(size_t)b * C + c] = l[c] - m - ls;
}

// ---------------------------------------------------------------------------
// Host driver
// ---------------------------------------------------------------------------
static inline int cdiv(long long a, int b) { return (int)((a + b - 1) / b); }

static inline void launch_gemm(const float* A, const _Float16* Bp, const float* bias,
                               const float* gamma, const float* beta, float* out,
                               int M, int Nout, int Kpad, int fuse, hipStream_t s) {
  dim3 grid((Nout + 15) / 16, (M + 63) / 64);
  gemm_wmma<<<grid, 128, 0, s>>>(A, Bp, bias, gamma, beta, out, M, Nout, Kpad, fuse);
}

extern "C" void kernel_launch(void* const* d_in, const int* in_sizes, int n_in,
                              void* d_out, int out_size, void* d_ws, size_t ws_size,
                              hipStream_t stream) {
  (void)in_sizes; (void)n_in; (void)out_size; (void)ws_size;

  const int B = 16, N = 4096;
  const float* x = (const float*)d_in[0];

  // SA MLP layers: pytree dict keys sorted -> (W, b, beta, gamma) per layer.
  const float *saW[9], *saB[9], *saBeta[9], *saGamma[9];
  for (int l = 0; l < 9; ++l) {
    const int base = 1 + l * 4;
    saW[l]     = (const float*)d_in[base + 0];
    saB[l]     = (const float*)d_in[base + 1];
    saBeta[l]  = (const float*)d_in[base + 2];
    saGamma[l] = (const float*)d_in[base + 3];
  }
  // fc dict sorted: W1,W2,W3,b1,b2,b3,beta1,beta2,gamma1,gamma2
  const float* fcW[3]  = { (const float*)d_in[37], (const float*)d_in[38],
                           (const float*)d_in[39] };
  const float* fcB[3]  = { (const float*)d_in[40], (const float*)d_in[41],
                           (const float*)d_in[42] };
  const float* beta1   = (const float*)d_in[43];
  const float* beta2   = (const float*)d_in[44];
  const float* gam1    = (const float*)d_in[45];
  const float* gam2    = (const float*)d_in[46];

  // Layer shape table: 9 SA layers + 3 FC layers. (N=cout, K=cin)
  const int LN[12] = { 64, 64, 128, 128, 128, 256, 256, 512, 1024, 512, 256, 3 };
  const int LK[12] = {  3, 64,  64, 131, 128, 128, 259, 256,  512, 1024, 512, 256 };
  const float* LW[12] = { saW[0], saW[1], saW[2], saW[3], saW[4], saW[5],
                          saW[6], saW[7], saW[8], fcW[0], fcW[1], fcW[2] };

  // Workspace carving (256B aligned).
  char* ws = (char*)d_ws;
  size_t off = 0;
  auto carve = [&](size_t bytes) -> void* {
    void* p = ws + off;
    off = (off + bytes + 255) & ~(size_t)255;
    return p;
  };
  int*   idx1  = (int*)  carve((size_t)B * 512 * 4);
  float* nx1   = (float*)carve((size_t)B * 512 * 3 * 4);
  int*   gidx1 = (int*)  carve((size_t)B * 512 * 32 * 4);
  float* l1pts = (float*)carve((size_t)B * 512 * 128 * 4);
  int*   idx2  = (int*)  carve((size_t)B * 128 * 4);
  float* nx2   = (float*)carve((size_t)B * 128 * 3 * 4);
  int*   gidx2 = (int*)  carve((size_t)B * 128 * 64 * 4);
  float* l2pts = (float*)carve((size_t)B * 128 * 256 * 4);
  float* f3    = (float*)carve((size_t)B * 128 * 288 * 4);   // Kpad(259)=288
  float* l3    = (float*)carve((size_t)B * 1024 * 4);
  float* h1    = (float*)carve((size_t)B * 512 * 4);
  float* h2    = (float*)carve((size_t)B * 256 * 4);
  float* lg    = (float*)carve((size_t)B * 3 * 4);

  // Packed weight buffers (f16), per layer: Npad * Kpad halves.
  _Float16* Bp[12];
  for (int l = 0; l < 12; ++l) {
    const int N16  = (LN[l] + 15) / 16;
    const int Kpad = align32i(LK[l]);
    Bp[l] = (_Float16*)carve((size_t)N16 * 16 * Kpad * 2);
  }

  float* buf0 = (float*)carve((size_t)33554432 * 4);  // 128 MiB ping
  float* buf1 = (float*)carve((size_t)33554432 * 4);  // 128 MiB pong

  // ---- Pack all weights into WMMA B-fragment layout -----------------------
  for (int l = 0; l < 12; ++l) {
    const int N16 = (LN[l] + 15) / 16;
    const int KC  = align32i(LK[l]) >> 5;
    const long long total = (long long)N16 * KC * 32 * 16;
    pack_weights_kernel<<<cdiv(total, 256), 256, 0, stream>>>(
        LW[l], Bp[l], LN[l], LK[l], N16, KC);
  }

  // ---------------- SA1: 4096 -> 512, r=0.2, K=32, MLP [3,64,64,128] --------
  fps_kernel<<<B, 1024, 0, stream>>>(x, idx1, N, 512);
  gather_xyz_kernel<<<cdiv(B * 512, 256), 256, 0, stream>>>(x, idx1, nx1, N, 512, B * 512);
  query_ball_kernel<<<cdiv(B * 512, 256), 256, 0, stream>>>(x, nx1, gidx1, N, 512, 32,
                                                            0.2f * 0.2f, B * 512);
  group_feats_kernel<<<cdiv((long long)B * 512 * 32, 256), 256, 0, stream>>>(
      x, nx1, gidx1, nullptr, buf0, N, 512, 32, 0, 32, B * 512 * 32);
  const int M1 = B * 512 * 32;  // 262144
  launch_gemm(buf0, Bp[0], saB[0], saGamma[0], saBeta[0], buf1, M1,  64,  32, 1, stream);
  launch_gemm(buf1, Bp[1], saB[1], saGamma[1], saBeta[1], buf0, M1,  64,  64, 1, stream);
  launch_gemm(buf0, Bp[2], saB[2], saGamma[2], saBeta[2], buf1, M1, 128,  64, 1, stream);
  rowmax_kernel<<<cdiv((long long)B * 512 * 128, 256), 256, 0, stream>>>(
      buf1, l1pts, B * 512, 32, 128);

  // ---------------- SA2: 512 -> 128, r=0.4, K=64, MLP [131,128,128,256] -----
  fps_kernel<<<B, 1024, 0, stream>>>(nx1, idx2, 512, 128);
  gather_xyz_kernel<<<cdiv(B * 128, 256), 256, 0, stream>>>(nx1, idx2, nx2, 512, 128, B * 128);
  query_ball_kernel<<<cdiv(B * 128, 256), 256, 0, stream>>>(nx1, nx2, gidx2, 512, 128, 64,
                                                            0.4f * 0.4f, B * 128);
  group_feats_kernel<<<cdiv((long long)B * 128 * 64, 256), 256, 0, stream>>>(
      nx1, nx2, gidx2, l1pts, buf0, 512, 128, 64, 128, 160, B * 128 * 64);
  const int M2 = B * 128 * 64;  // 131072
  launch_gemm(buf0, Bp[3], saB[3], saGamma[3], saBeta[3], buf1, M2, 128, 160, 1, stream);
  launch_gemm(buf1, Bp[4], saB[4], saGamma[4], saBeta[4], buf0, M2, 128, 128, 1, stream);
  launch_gemm(buf0, Bp[5], saB[5], saGamma[5], saBeta[5], buf1, M2, 256, 128, 1, stream);
  rowmax_kernel<<<cdiv((long long)B * 128 * 256, 256), 256, 0, stream>>>(
      buf1, l2pts, B * 128, 64, 256);

  // ---------------- SA3 (global): MLP [259,256,512,1024], max over 128 ------
  concat3_kernel<<<cdiv(B * 128, 256), 256, 0, stream>>>(nx2, l2pts, f3, B * 128, 256, 288);
  const int M3 = B * 128;  // 2048
  launch_gemm(f3,   Bp[6], saB[6], saGamma[6], saBeta[6], buf0, M3,  256, 288, 1, stream);
  launch_gemm(buf0, Bp[7], saB[7], saGamma[7], saBeta[7], buf1, M3,  512, 256, 1, stream);
  launch_gemm(buf1, Bp[8], saB[8], saGamma[8], saBeta[8], buf0, M3, 1024, 512, 1, stream);
  rowmax_kernel<<<cdiv((long long)B * 1024, 256), 256, 0, stream>>>(buf0, l3, B, 128, 1024);

  // ---------------- FC head + log_softmax -----------------------------------
  launch_gemm(l3, Bp[9],  fcB[0], gam1, beta1, h1, B, 512, 1024, 1, stream);
  launch_gemm(h1, Bp[10], fcB[1], gam2, beta2, h2, B, 256,  512, 1, stream);
  launch_gemm(h2, Bp[11], fcB[2], nullptr, nullptr, lg, B, 3, 256, 0, stream);
  logsoftmax_kernel<<<1, 32, 0, stream>>>(lg, (float*)d_out, B, 3);
}
